// DSNN_53343493816852
// MI455X (gfx1250) — matrix operand order repaired and checked by reading, not compiled, not run on hardware
//
#include <hip/hip_runtime.h>
#include <stdint.h>

typedef __attribute__((ext_vector_type(16))) _Float16 v16h;
typedef __attribute__((ext_vector_type(8)))  float    v8f;
typedef __attribute__((ext_vector_type(4)))  int      v4i;

#define GLOBAL_AS __attribute__((address_space(1)))

#define ALPHA 0.9f
#define BETA  0.85f

// ---- LDS layout (units: halves), rows padded to decorrelate banks ----
#define SIN_STRIDE 520            // 512 + 8
#define OUT_STRIDE 1032           // 1024 + 8
#define SIN_BYTES  (16 * SIN_STRIDE * 2)          // 16640
#define OUT_BYTES  (16 * OUT_STRIDE * 2)          // 33024
#define SMEM_BYTES (SIN_BYTES + 2 * OUT_BYTES)    // 82688

// ---- workspace layout (bytes) ----
#define WS_W0T 0                         // half [1024][512]
#define WS_B0  1048576                   // float[1024]
#define WS_W1T 1052672                   // half [1024][1024]
#define WS_W2T 3149824                   // half [512][1024]

__device__ __forceinline__ v8f wmma16(v16h a, v16h b, v8f c) {
  return __builtin_amdgcn_wmma_f32_16x16x32_f16(false, a, false, b, (short)0, c,
                                                false, false);
}

// Opaque pointer pass-through (defeats LICM hoist-and-spill of the weight
// loads across the time loop) that ALSO re-asserts the global address space,
// so weight loads lower to global_load_b128 (saddr + 32-bit voffset, LOADcnt
// only) instead of flat_load_b128 (64-bit vaddr, LOADcnt+DScnt+XCNT).
__device__ __forceinline__ const GLOBAL_AS _Float16* launder_g(const _Float16* p) {
  asm volatile("" : "+s"(p));
  return (const GLOBAL_AS _Float16*)(uintptr_t)p;
}

// A fragment (16x32 f16) from LDS, row-major with padded stride.
// lanes 0-15: row=lane,   K = k0+0..15
// lanes16-31: row=lane-16,K = k0+16..31
__device__ __forceinline__ v16h load_a_lds(const _Float16* base, int strideH,
                                           int lane, int k0) {
  const int row = lane & 15;
  const int ks  = (lane >> 4) * 16;
  const _Float16* p = base + row * strideH + k0 + ks;
  union { v4i q[2]; v16h h; } u;
  u.q[0] = *(const v4i*)(p);        // ds_load_b128
  u.q[1] = *(const v4i*)(p + 8);    // +16 bytes
  return u.h;
}

// B fragment (32x16 f16) from transposed weights Wt[n][k] in global memory.
// lanes 0-15: col = n0+lane,   K = k0..k0+15   (32 contiguous bytes)
// lanes16-31: col = n0+lane-16,K = k0+16..k0+31
__device__ __forceinline__ v16h load_b_glb(const GLOBAL_AS _Float16* wt,
                                           int strideH, int lane, int n0, int k0) {
  const int col = n0 + (lane & 15);
  const int ks  = (lane >> 4) * 16;
  return *(const GLOBAL_AS v16h*)(wt + (size_t)col * strideH + k0 + ks);
}

// ---------------- weight convert + transpose (once per call) ----------------
__global__ void snn_convert(const float* __restrict__ W0,
                            const float* __restrict__ W1,
                            const float* __restrict__ W2,
                            _Float16* __restrict__ w0t, float* __restrict__ b0v,
                            _Float16* __restrict__ w1t, _Float16* __restrict__ w2t) {
  const int NT0 = 1024 * 512, NB = 1024, NT1 = 1024 * 1024, NT2 = 512 * 1024;
  const int total = NT0 + NB + NT1 + NT2;
  for (int i = blockIdx.x * blockDim.x + threadIdx.x; i < total;
       i += gridDim.x * blockDim.x) {
    int idx = i;
    if (idx < NT0) {                       // W0t[n][k] = W0[k][n], k<512
      int n = idx >> 9, k = idx & 511;
      w0t[idx] = (_Float16)W0[(size_t)k * 1024 + n];
      continue;
    }
    idx -= NT0;
    if (idx < NB) {                        // bias row 512 of W0, kept fp32
      b0v[idx] = W0[(size_t)512 * 1024 + idx];
      continue;
    }
    idx -= NB;
    if (idx < NT1) {                       // W1t[n][k] = W1[k][n]
      int n = idx >> 10, k = idx & 1023;
      w1t[idx] = (_Float16)W1[(size_t)k * 1024 + n];
      continue;
    }
    idx -= NT1;
    {                                      // W2t[n][k] = W2[k][n]
      int n = idx >> 10, k = idx & 1023;
      w2t[idx] = (_Float16)W2[(size_t)k * 512 + n];
    }
  }
}

// ---------------- persistent SNN kernel: 1 WG per 16-row batch tile ---------
__global__ void __launch_bounds__(512) snn_main(
    const float* __restrict__ inputs, const float* __restrict__ rnd,
    const _Float16* __restrict__ w0t, const float* __restrict__ b0v,
    const _Float16* __restrict__ w1t, const _Float16* __restrict__ w2t,
    float* __restrict__ out) {
  extern __shared__ char smem[];
  _Float16* s_in = (_Float16*)smem;                       // [16][520]
  _Float16* s_o0 = (_Float16*)(smem + SIN_BYTES);         // [16][1032]
  _Float16* s_o1 = (_Float16*)(smem + SIN_BYTES + OUT_BYTES);

  const int tid  = threadIdx.x;            // 512 threads = 16 waves
  const int lane = tid & 31;
  const int wave = tid >> 5;               // 0..15
  const int gb0  = blockIdx.x * 16;        // batch tile base row
  const int hs   = lane >> 4;              // half-lane select
  const int l15  = lane & 15;

  // N-tile assignment: layer0/1: 4 tiles/wave, layer2: 2 tiles/wave
  int n01[4], n2t[2];
#pragma unroll
  for (int j = 0; j < 4; ++j) n01[j] = (wave * 4 + j) * 16;
#pragma unroll
  for (int j = 0; j < 2; ++j) n2t[j] = (wave * 2 + j) * 16;

  // per-lane bias (C-fragment: column fixed per lane across all 8 VGPRs)
  float bias[4];
#pragma unroll
  for (int j = 0; j < 4; ++j) bias[j] = b0v[n01[j] + l15];

  // time-invariant input column for this thread: hoist 16 loads out of t-loop
  const int colA = tid & 511;
  float xin[16];
#pragma unroll
  for (int r = 0; r < 16; ++r)
    xin[r] = inputs[((size_t)(gb0 + r)) * 512 + colA];

  // persistent state in C-fragment registers
  v8f mem0[4], syn1[4], mem1[4], syn2[2], mem2[2];
#pragma unroll
  for (int j = 0; j < 4; ++j) { mem0[j] = (v8f)0.0f; syn1[j] = (v8f)0.0f; mem1[j] = (v8f)0.0f; }
#pragma unroll
  for (int j = 0; j < 2; ++j) { syn2[j] = (v8f)0.0f; mem2[j] = (v8f)0.0f; }

  const _Float16 one_h  = (_Float16)1.0f;
  const _Float16 zero_h = (_Float16)0.0f;

  for (int t = 1; t < 100; ++t) {
    // Re-launder weight bases every step: loads stay inside the loop and
    // stream from the (L2-resident) transposed weight images.
    const GLOBAL_AS _Float16* w0 = launder_g(w0t);
    const GLOBAL_AS _Float16* w1 = launder_g(w1t);
    const GLOBAL_AS _Float16* w2 = launder_g(w2t);

    // ---- phase A: Poisson spike encoding of inputs into LDS ----
    {
      const float* rp = rnd + ((size_t)t * 256 + gb0) * 513 + colA;
#pragma unroll 4
      for (int r = 0; r < 16; ++r) {
        float rv = rp[(size_t)r * 513];
        s_in[r * SIN_STRIDE + colA] = (xin[r] > rv) ? one_h : zero_h;
      }
    }
    __syncthreads();

    // ---- phase B: layer0  syn0 = S@W0 + b0 (full replace) ----
    {
      v8f acc[4];
#pragma unroll
      for (int j = 0; j < 4; ++j) acc[j] = (v8f)0.0f;
#pragma unroll 1
      for (int k0 = 0; k0 < 512; k0 += 32) {
        v16h a = load_a_lds(s_in, SIN_STRIDE, lane, k0);
#pragma unroll
        for (int j = 0; j < 4; ++j) {
          v16h b = load_b_glb(w0, 512, lane, n01[j], k0);
          acc[j] = wmma16(a, b, acc[j]);
        }
      }
#pragma unroll
      for (int j = 0; j < 4; ++j) {
#pragma unroll
        for (int v = 0; v < 8; ++v) {
          float m  = BETA * mem0[j][v] + acc[j][v] + bias[j];
          bool  sp = (m - 1.0f) > 0.0f;
          s_o0[(hs * 8 + v) * OUT_STRIDE + n01[j] + l15] = sp ? one_h : zero_h;
          mem0[j][v] = sp ? 0.0f : m;
        }
      }
    }
    __syncthreads();

    // ---- phase C: layer1  syn1 = a*syn1 + out0@W1 ----
    {
      v8f acc[4];
#pragma unroll
      for (int j = 0; j < 4; ++j) acc[j] = (v8f)0.0f;
#pragma unroll 1
      for (int k0 = 0; k0 < 1024; k0 += 32) {
        v16h a = load_a_lds(s_o0, OUT_STRIDE, lane, k0);
#pragma unroll
        for (int j = 0; j < 4; ++j) {
          v16h b = load_b_glb(w1, 1024, lane, n01[j], k0);
          acc[j] = wmma16(a, b, acc[j]);
        }
      }
#pragma unroll
      for (int j = 0; j < 4; ++j) {
#pragma unroll
        for (int v = 0; v < 8; ++v) {
          float s = ALPHA * syn1[j][v] + acc[j][v];
          syn1[j][v] = s;
          float m  = BETA * mem1[j][v] + s;
          bool  sp = (m - 1.0f) > 0.0f;
          s_o1[(hs * 8 + v) * OUT_STRIDE + n01[j] + l15] = sp ? one_h : zero_h;
          mem1[j][v] = sp ? 0.0f : m;
        }
      }
    }
    __syncthreads();

    // ---- phase D: layer2  syn2 = a*syn2 + out1@W2 ; no spike/reset ----
    {
      v8f acc[2];
#pragma unroll
      for (int j = 0; j < 2; ++j) acc[j] = (v8f)0.0f;
#pragma unroll 1
      for (int k0 = 0; k0 < 1024; k0 += 32) {
        v16h a = load_a_lds(s_o1, OUT_STRIDE, lane, k0);
#pragma unroll
        for (int j = 0; j < 2; ++j) {
          v16h b = load_b_glb(w2, 1024, lane, n2t[j], k0);
          acc[j] = wmma16(a, b, acc[j]);
        }
      }
#pragma unroll
      for (int j = 0; j < 2; ++j) {
#pragma unroll
        for (int v = 0; v < 8; ++v) {
          float s = ALPHA * syn2[j][v] + acc[j][v];
          syn2[j][v] = s;
          mem2[j][v] = BETA * mem2[j][v] + s;
        }
      }
    }
    __syncthreads();
  }

  // ---- final: store mem2 fragments -> out[256][512] f32 ----
#pragma unroll
  for (int j = 0; j < 2; ++j) {
#pragma unroll
    for (int v = 0; v < 8; ++v) {
      int row = hs * 8 + v;
      out[(size_t)(gb0 + row) * 512 + n2t[j] + l15] = mem2[j][v];
    }
  }
}

extern "C" void kernel_launch(void* const* d_in, const int* in_sizes, int n_in,
                              void* d_out, int out_size, void* d_ws, size_t ws_size,
                              hipStream_t stream) {
  const float* inputs = (const float*)d_in[0];
  const float* W0     = (const float*)d_in[1];   // [513,1024]
  const float* W1     = (const float*)d_in[2];   // [1024,1024]
  const float* W2     = (const float*)d_in[3];   // [1024,512]
  const float* rnd    = (const float*)d_in[4];   // [100,256,513]
  float* out          = (float*)d_out;           // [256,512]

  char* ws = (char*)d_ws;
  _Float16* w0t = (_Float16*)(ws + WS_W0T);
  float*    b0v = (float*)   (ws + WS_B0);
  _Float16* w1t = (_Float16*)(ws + WS_W1T);
  _Float16* w2t = (_Float16*)(ws + WS_W2T);

  snn_convert<<<512, 256, 0, stream>>>(W0, W1, W2, w0t, b0v, w1t, w2t);

  hipFuncSetAttribute((const void*)snn_main,
                      hipFuncAttributeMaxDynamicSharedMemorySize, SMEM_BYTES);
  snn_main<<<16, 512, SMEM_BYTES, stream>>>(inputs, rnd, w0t, b0v, w1t, w2t, out);
}
